// _MultiHeadAttention_29686813949966
// MI455X (gfx1250) — matrix operand order repaired and checked by reading, not compiled, not run on hardware
//
#include <hip/hip_runtime.h>
#include <hip/hip_bf16.h>

typedef __attribute__((ext_vector_type(16))) _Float16 v16h;
typedef __attribute__((ext_vector_type(8)))  float    v8f;

#define Bq   4
#define Tq   1024
#define Cq   256
#define Hq   8
#define HDq  32
#define Wq   64
#define TG   1088            // 64 pad rows (g = t + 64) + 1024 real rows
#define BT   (Bq*Tq)         // 4096

// ---------------- helpers ----------------

// A-fragment (16x32 f16, MxK): lane row = lane&15; hi=lane>>4 selects K-halves.
// 16 contiguous K values K(kb+hi*16 .. +15): two 16B chunks at p, p+8.
__device__ __forceinline__ v16h load_frag_a(const _Float16* p) {
    v16h a;
    ((float4*)&a)[0] = *(const float4*)(p);
    ((float4*)&a)[1] = *(const float4*)(p + 8);
    return a;
}
// B-fragment (32x16 f16, KxN): lane col = lane&15; lanes 0-15 K=0..15, 16-31 K=16..31.
// 16 contiguous K values: two 16B chunks at p, p+8.
__device__ __forceinline__ v16h load_frag_b(const _Float16* p) {
    v16h b;
    ((float4*)&b)[0] = *(const float4*)(p);
    ((float4*)&b)[1] = *(const float4*)(p + 8);
    return b;
}

// ---------------- kernels ----------------

__global__ void cvt_f32_f16(const float* __restrict__ src, _Float16* __restrict__ dst, int n) {
    int i = blockIdx.x * blockDim.x + threadIdx.x;
    if (i < n) dst[i] = (_Float16)src[i];
}

// Fill window-pad rows g=0..63: k rows get bk bias, v (transposed) gets bv bias.
__global__ void pad_init(const float* __restrict__ b_in,
                         _Float16* __restrict__ Kpad, _Float16* __restrict__ Vt) {
    int i = blockIdx.x * blockDim.x + threadIdx.x;   // 4*64*256 = 65536
    int b = i >> 14, g = (i >> 8) & 63, c = i & 255;
    Kpad[((b * TG) + g) * Cq + c]   = (_Float16)b_in[Cq + c];
    Vt[((b * Cq) + c) * TG + g]     = (_Float16)b_in[2 * Cq + c];
}

// Y = X @ Wqkv^T + b : one wave per 16x16 tile; epilogue routes to Q / Kpad / Vt.
__global__ void gemm_qkv(const _Float16* __restrict__ X, const _Float16* __restrict__ Wh,
                         const float* __restrict__ bias,
                         _Float16* __restrict__ Q, _Float16* __restrict__ Kpad,
                         _Float16* __restrict__ Vt) {
    int wid  = blockIdx.x * (blockDim.x >> 5) + (threadIdx.x >> 5);
    int lane = threadIdx.x & 31;
    int nt = wid % 48, mt = wid / 48;              // 48 N-tiles (768), 256 M-tiles
    int row0 = mt * 16, n0 = nt * 16;
    int nl = lane & 15, hi = lane >> 4;

    const _Float16* arow = X  + (row0 + nl) * Cq;
    const _Float16* brow = Wh + (n0   + nl) * Cq;
    v8f acc = {};
    #pragma unroll
    for (int k = 0; k < Cq; k += 32) {
        v16h a = load_frag_a(arow + k + hi * 16);
        v16h b = load_frag_b(brow + k + hi * 16);
        acc = __builtin_amdgcn_wmma_f32_16x16x32_f16(false, a, false, b,
                                                     (short)0, acc, false, false);
    }
    float bv  = bias[n0 + nl];
    int   sec = n0 >> 8;                 // 0:Q 1:K 2:V
    int   c   = (n0 & 255) + nl;
    #pragma unroll
    for (int i = 0; i < 8; ++i) {
        int rowg = row0 + i + hi * 8;
        int b_   = rowg >> 10, t = rowg & 1023;
        _Float16 h = (_Float16)(acc[i] + bv);
        if      (sec == 0) Q[rowg * Cq + c] = h;
        else if (sec == 1) Kpad[((b_ * TG) + t + 64) * Cq + c] = h;
        else               Vt[((b_ * Cq) + c) * TG + (t + 64)] = h;
    }
}

// Sliding-window attention: one wave per (b, h, 16-query tile).
__global__ void attn(const _Float16* __restrict__ Q, const _Float16* __restrict__ Kpad,
                     const _Float16* __restrict__ Vt, _Float16* __restrict__ Ctx) {
    __shared__ _Float16 lds[4 * 16 * 96];
    int lwave = threadIdx.x >> 5;
    int wid   = blockIdx.x * 4 + lwave;
    int lane  = threadIdx.x & 31;
    int tile = wid & 63, h = (wid >> 6) & 7, b = wid >> 9;
    int t0 = tile * 16;
    int nl = lane & 15, hi = lane >> 4;
    _Float16* P = lds + lwave * 16 * 96;

    // Q A-fragment (16 queries x HD=32)
    const _Float16* qrow = Q + (((b << 10) + t0 + nl) * Cq) + h * HDq;
    v16h qa = load_frag_a(qrow + hi * 16);

    // scores: 5 key tiles of 16 abs positions; padded row g = t0 + col (col = j*16+nl)
    v8f s[5];
    #pragma unroll
    for (int j = 0; j < 5; ++j) {
        const _Float16* krow = Kpad + ((b * TG) + t0 + j * 16 + nl) * Cq + h * HDq + hi * 16;
        v16h kb = load_frag_b(krow);
        v8f z = {};
        s[j] = __builtin_amdgcn_wmma_f32_16x16x32_f16(false, qa, false, kb,
                                                      (short)0, z, false, false);
    }

    // masked scaled softmax over the 80 columns (64 valid per row)
    const float scale = 0.17677669529663687f;   // 1/sqrt(32)
    #pragma unroll
    for (int i = 0; i < 8; ++i) {
        int m = i + hi * 8;                     // C-layout: row m in this lane-half
        float mx = -3.0e38f;
        #pragma unroll
        for (int j = 0; j < 5; ++j) {
            int d = m - (j * 16 + nl - 64);     // window distance
            bool valid = (d >= 0) && (d < Wq);
            float v = valid ? s[j][i] * scale : -3.0e38f;
            s[j][i] = v;
            mx = fmaxf(mx, v);
        }
        #pragma unroll
        for (int msk = 1; msk < 16; msk <<= 1) mx = fmaxf(mx, __shfl_xor(mx, msk, 32));
        float sum = 0.f;
        #pragma unroll
        for (int j = 0; j < 5; ++j) {
            float e = __expf(s[j][i] - mx);
            s[j][i] = e; sum += e;
        }
        #pragma unroll
        for (int msk = 1; msk < 16; msk <<= 1) sum += __shfl_xor(sum, msk, 32);
        float inv = 1.0f / sum;
        #pragma unroll
        for (int j = 0; j < 5; ++j)
            P[m * 96 + j * 16 + nl] = (_Float16)(s[j][i] * inv);   // C-layout -> LDS
    }
    // zero columns 80..95 (K padding for the 3rd 32-wide ctx chunk)
    *(float4*)(P + nl * 96 + 80 + hi * 8) = make_float4(0.f, 0.f, 0.f, 0.f);

    // re-load P as A-fragments (LDS transpose), 3 chunks of K=32
    v16h pa[3];
    #pragma unroll
    for (int kc = 0; kc < 3; ++kc)
        pa[kc] = load_frag_a(P + nl * 96 + kc * 32 + hi * 16);

    // ctx = P @ V : N = HD (2 tiles), K = 96 (3 chunks); Vt pos = t0 + col
    #pragma unroll
    for (int ntile = 0; ntile < 2; ++ntile) {
        int c = h * HDq + ntile * 16 + nl;
        v8f acc = {};
        #pragma unroll
        for (int kc = 0; kc < 3; ++kc) {
            bool oob = (kc == 2) && hi;         // cols 80..95: P==0, avoid OOB read
            const _Float16* vp = Vt + ((b * Cq) + c) * TG + (oob ? 0 : (t0 + kc * 32 + hi * 16));
            v16h vb = load_frag_b(vp);
            if (oob) { v16h z = {}; vb = z; }
            acc = __builtin_amdgcn_wmma_f32_16x16x32_f16(false, pa[kc], false, vb,
                                                         (short)0, acc, false, false);
        }
        #pragma unroll
        for (int i = 0; i < 8; ++i) {
            int rowg = (b << 10) + t0 + i + hi * 8;
            Ctx[rowg * Cq + c] = (_Float16)acc[i];
        }
    }
}

// Out = Ctx @ Wout^T + b, f32 store to d_out
__global__ void gemm_out(const _Float16* __restrict__ Ctx, const _Float16* __restrict__ Woh,
                         const float* __restrict__ bias, float* __restrict__ out) {
    int wid  = blockIdx.x * (blockDim.x >> 5) + (threadIdx.x >> 5);
    int lane = threadIdx.x & 31;
    int nt = wid & 15, mt = wid >> 4;             // 16 N-tiles, 256 M-tiles
    int row0 = mt * 16, n0 = nt * 16;
    int nl = lane & 15, hi = lane >> 4;

    const _Float16* arow = Ctx + (row0 + nl) * Cq;
    const _Float16* brow = Woh + (n0   + nl) * Cq;
    v8f acc = {};
    #pragma unroll
    for (int k = 0; k < Cq; k += 32) {
        v16h a = load_frag_a(arow + k + hi * 16);
        v16h b = load_frag_b(brow + k + hi * 16);
        acc = __builtin_amdgcn_wmma_f32_16x16x32_f16(false, a, false, b,
                                                     (short)0, acc, false, false);
    }
    float bv = bias[n0 + nl];
    #pragma unroll
    for (int i = 0; i < 8; ++i) {
        int rowg = row0 + i + hi * 8;
        out[rowg * Cq + n0 + nl] = acc[i] + bv;
    }
}

// ---------------- launch ----------------

extern "C" void kernel_launch(void* const* d_in, const int* in_sizes, int n_in,
                              void* d_out, int out_size, void* d_ws, size_t ws_size,
                              hipStream_t stream) {
    const float* inputs = (const float*)d_in[0];   // (4,1024,256)
    const float* w_in   = (const float*)d_in[1];   // (768,256)
    const float* b_in   = (const float*)d_in[2];   // (768,)
    const float* w_out  = (const float*)d_in[3];   // (256,256)
    const float* b_out  = (const float*)d_in[4];   // (256,)
    float* out = (float*)d_out;

    // workspace layout (f16 elements), ~10.8 MB total
    _Float16* p    = (_Float16*)d_ws;
    _Float16* Xh   = p;                       p += BT * Cq;          // 1048576
    _Float16* Wh   = p;                       p += 3 * Cq * Cq;      // 196608
    _Float16* Woh  = p;                       p += Cq * Cq;          // 65536
    _Float16* Qh   = p;                       p += BT * Cq;          // 1048576
    _Float16* Kpad = p;                       p += Bq * TG * Cq;     // 1114112
    _Float16* Vt   = p;                       p += Bq * Cq * TG;     // 1114112
    _Float16* Ctx  = p;                       p += BT * Cq;          // 1048576

    cvt_f32_f16<<<(BT * Cq) / 256, 256, 0, stream>>>(inputs, Xh, BT * Cq);
    cvt_f32_f16<<<(3 * Cq * Cq) / 256, 256, 0, stream>>>(w_in, Wh, 3 * Cq * Cq);
    cvt_f32_f16<<<(Cq * Cq) / 256, 256, 0, stream>>>(w_out, Woh, Cq * Cq);
    pad_init<<<(Bq * 64 * Cq) / 256, 256, 0, stream>>>(b_in, Kpad, Vt);

    // QKV: 256 M-tiles x 48 N-tiles = 12288 waves, 8 waves/block
    gemm_qkv<<<1536, 256, 0, stream>>>(Xh, Wh, b_in, Qh, Kpad, Vt);

    // attention: 4*8*64 = 2048 waves, 4 waves/block
    attn<<<512, 128, 0, stream>>>(Qh, Kpad, Vt, Ctx);

    // out projection: 256 x 16 tiles = 4096 waves, 8 waves/block
    gemm_out<<<512, 256, 0, stream>>>(Ctx, Woh, b_out, out);
}
